// SpatialBlock_20057497272920
// MI455X (gfx1250) — matrix operand order repaired and checked by reading, compiled, not run on hardware
//
#include <hip/hip_runtime.h>
#include <hip/hip_bf16.h>

// ---------------- problem constants (match reference) ----------------
#define NT 4000
#define NS 5000
#define BD 128          // B * D flattened columns
#define DD 64           // D
#define RR 30           // low-rank dim
#define KPS 5024        // Ns rounded up to 32
#define KPT 4000        // Nt already multiple of 32

typedef __bf16 v16bf __attribute__((ext_vector_type(16)));
typedef float  v8f   __attribute__((ext_vector_type(8)));

union FragB { unsigned u[8]; v16bf v; };

__device__ __forceinline__ unsigned short f2bf(float f) {
    union { float f; unsigned u; } x; x.f = f;
    unsigned r = x.u + 0x7FFFu + ((x.u >> 16) & 1u);   // round-to-nearest-even
    return (unsigned short)(r >> 16);
}
__device__ __forceinline__ unsigned packbf(float a, float b) {
    return (unsigned)f2bf(a) | ((unsigned)f2bf(b) << 16);
}

// ---------------------------------------------------------------------
// adp = softmax(relu(nv1 @ nv2), axis=1), written as bf16.  One block/row.
// Scores recomputed per pass (2*R flops/elt) -> no fp32 score buffer.
// ---------------------------------------------------------------------
__global__ __launch_bounds__(256)
void adp_kernel(const float* __restrict__ nv1, const float* __restrict__ nv2,
                unsigned short* __restrict__ out, int n, int m) {
    __shared__ float v1[32];
    __shared__ float red[256];
    const int row = blockIdx.x, tid = threadIdx.x;
    if (tid < RR) v1[tid] = nv1[row * RR + tid];
    __syncthreads();

    float lmax = 0.f;
    for (int c = tid; c < m; c += 256) {
        float s = 0.f;
        #pragma unroll
        for (int r = 0; r < RR; ++r) s = fmaf(v1[r], nv2[(long)r * m + c], s);
        lmax = fmaxf(lmax, s);
    }
    red[tid] = lmax; __syncthreads();
    for (int off = 128; off > 0; off >>= 1) {
        if (tid < off) red[tid] = fmaxf(red[tid], red[tid + off]);
        __syncthreads();
    }
    const float rmax = red[0];
    __syncthreads();

    float lsum = 0.f;
    for (int c = tid; c < m; c += 256) {
        float s = 0.f;
        #pragma unroll
        for (int r = 0; r < RR; ++r) s = fmaf(v1[r], nv2[(long)r * m + c], s);
        s = fmaxf(s, 0.f);
        lsum += expf(s - rmax);
    }
    red[tid] = lsum; __syncthreads();
    for (int off = 128; off > 0; off >>= 1) {
        if (tid < off) red[tid] += red[tid + off];
        __syncthreads();
    }
    const float inv = 1.f / red[0];
    __syncthreads();

    for (int c = tid; c < m; c += 256) {
        float s = 0.f;
        #pragma unroll
        for (int r = 0; r < RR; ++r) s = fmaf(v1[r], nv2[(long)r * m + c], s);
        s = fmaxf(s, 0.f);
        out[(long)row * m + c] = f2bf(expf(s - rmax) * inv);
    }
}

// ---------------------------------------------------------------------
// Pack x (B,m,D) fp32 -> XT (128 x Kp) bf16, XT[(b*64+d)*Kp + mm] = x[b,mm,d]
// ---------------------------------------------------------------------
__global__ __launch_bounds__(256)
void pack_xt(const float* __restrict__ x, unsigned short* __restrict__ XT,
             int m, int Kp) {
    const long total = 128L * Kp;
    for (long idx = (long)blockIdx.x * blockDim.x + threadIdx.x; idx < total;
         idx += (long)gridDim.x * blockDim.x) {
        int j  = (int)(idx / Kp);
        long mm = idx - (long)j * Kp;
        float v = (mm < m) ? x[(long)(j >> 6) * m * DD + mm * DD + (j & 63)] : 0.f;
        XT[idx] = f2bf(v);
    }
}

__global__ __launch_bounds__(256)
void copyf(const float* __restrict__ a, float* __restrict__ o, long nel) {
    for (long i = (long)blockIdx.x * blockDim.x + threadIdx.x; i < nel;
         i += (long)gridDim.x * blockDim.x) o[i] = a[i];
}

// ---------------------------------------------------------------------
// C(n x 128, fp32, stride ldc, col offset colOff) = A(n x m) @ X
// X given transposed as XT(128 x Kp) bf16.  A fp32 (converted to bf16
// while staging) or bf16.  Block tile 64x128, 8 waves (4 row x 2 col),
// v_wmma_f32_16x16x32_bf16.  Double-buffered LDS, software pipelined:
//   regs <- global tile kt+1;  wmma from LDS[kt&1];  LDS[nxt] <- regs; barrier
// ---------------------------------------------------------------------
template <bool AF32>
__global__ __launch_bounds__(256)
void gemm_bf16(const void* __restrict__ Aptr, const unsigned short* __restrict__ XT,
               float* __restrict__ C, int n, int m, int Kp, int ldc, int colOff) {
    __shared__ __align__(16) unsigned As[2][64 * 16];    // 2 x 4 KB
    __shared__ __align__(16) unsigned Bs[2][128 * 16];   // 2 x 8 KB

    const int tid  = threadIdx.x;
    const int lane = tid & 31, wave = tid >> 5;
    const int rw = wave >> 1, cw = wave & 1;
    const int half = lane >> 4, l16 = lane & 15;
    const int rowBase = blockIdx.x * 64;
    const bool fullRows = (rowBase + 64 <= n);

    // staging assignment: contiguous pairs per thread (b128-friendly)
    const int rA = tid >> 2, cA = (tid & 3) << 2;   // A: row 0..63, pair base
    const int rB = tid >> 1, cB = (tid & 1) << 3;   // B: row 0..127, pair base
    unsigned aReg[4], bReg[8];

    auto loadTiles = [&](int k0) {
        const int gr = rowBase + rA;
        const int kA = k0 + (cA << 1);
        if (fullRows && (k0 + 32 <= m)) {
            if (AF32) {
                const float4* p = (const float4*)((const float*)Aptr + (long)gr * m + kA);
                float4 v0 = p[0], v1 = p[1];
                aReg[0] = packbf(v0.x, v0.y); aReg[1] = packbf(v0.z, v0.w);
                aReg[2] = packbf(v1.x, v1.y); aReg[3] = packbf(v1.z, v1.w);
            } else {
                uint4 v = *(const uint4*)((const unsigned short*)Aptr + (long)gr * m + kA);
                aReg[0] = v.x; aReg[1] = v.y; aReg[2] = v.z; aReg[3] = v.w;
            }
        } else {
            #pragma unroll
            for (int i = 0; i < 4; ++i) {
                int k = kA + (i << 1);
                unsigned val = 0u;
                if (gr < n) {
                    if (AF32) {
                        const float* Af = (const float*)Aptr;
                        float a0 = (k     < m) ? Af[(long)gr * m + k]     : 0.f;
                        float a1 = (k + 1 < m) ? Af[(long)gr * m + k + 1] : 0.f;
                        val = packbf(a0, a1);
                    } else {
                        const unsigned short* Ab = (const unsigned short*)Aptr;
                        unsigned a0 = (k     < m) ? (unsigned)Ab[(long)gr * m + k]     : 0u;
                        unsigned a1 = (k + 1 < m) ? (unsigned)Ab[(long)gr * m + k + 1] : 0u;
                        val = a0 | (a1 << 16);
                    }
                }
                aReg[i] = val;
            }
        }
        // B always in-bounds (XT is K-padded)
        const uint4* pb = (const uint4*)(XT + (long)rB * Kp + k0 + (cB << 1));
        uint4 w0 = pb[0], w1 = pb[1];
        bReg[0] = w0.x; bReg[1] = w0.y; bReg[2] = w0.z; bReg[3] = w0.w;
        bReg[4] = w1.x; bReg[5] = w1.y; bReg[6] = w1.z; bReg[7] = w1.w;
    };
    auto storeTiles = [&](int buf) {
        uint4* pa = (uint4*)&As[buf][(rA << 4) + cA];
        pa[0] = make_uint4(aReg[0], aReg[1], aReg[2], aReg[3]);
        uint4* pbs = (uint4*)&Bs[buf][(rB << 4) + cB];
        pbs[0] = make_uint4(bReg[0], bReg[1], bReg[2], bReg[3]);
        pbs[1] = make_uint4(bReg[4], bReg[5], bReg[6], bReg[7]);
    };

    v8f acc[4];
    #pragma unroll
    for (int t = 0; t < 4; ++t)
        #pragma unroll
        for (int e = 0; e < 8; ++e) acc[t][e] = 0.f;

    const int ksteps = (m + 31) >> 5;

    // prologue: stage tile 0
    loadTiles(0);
    storeTiles(0);
    __syncthreads();

    for (int kt = 0; kt < ksteps; ++kt) {
        const int buf = kt & 1;
        const bool more = (kt + 1) < ksteps;
        if (more) {
            loadTiles((kt + 1) << 5);              // hide global latency behind WMMA
            const int kp = (kt + 2) << 5;          // prefetch K+2 A-tile
            if (kp < m && fullRows) {
                if (AF32)
                    __builtin_prefetch((const float*)Aptr + (long)(rowBase + rA) * m + kp, 0, 1);
                else
                    __builtin_prefetch((const unsigned short*)Aptr + (long)(rowBase + rA) * m + kp, 0, 1);
            }
        }

        // fragments per ISA 7.12.2 layouts
        FragB af;
        const int arow = rw * 16 + l16;
        #pragma unroll
        for (int j = 0; j < 8; ++j) {
            // A 16x32 bf16: VGPR j (grp g=j>>2): K = g*16 + half*8 + 2*(j&3)
            int pj = ((j >> 2) << 3) + (half << 2) + (j & 3);
            af.u[j] = As[buf][arow * 16 + pj];
        }
        #pragma unroll
        for (int t = 0; t < 4; ++t) {
            FragB bfg;
            const int ncol = cw * 64 + t * 16 + l16;
            #pragma unroll
            for (int j = 0; j < 8; ++j)            // B 32x16 bf16: VGPR j (grp g=j>>2): K = g*16 + half*8 + 2*(j&3)
                bfg.u[j] = Bs[buf][ncol * 16 + ((j >> 2) << 3) + (half << 2) + (j & 3)];
            acc[t] = __builtin_amdgcn_wmma_f32_16x16x32_bf16(
                false, af.v, false, bfg.v, (short)0, acc[t], false, false);
        }

        if (more) {
            storeTiles(buf ^ 1);                   // other buffer: no read/write hazard
            __syncthreads();                       // single barrier per K-step
        }
    }

    // store D: VGPR v holds M = half*8 + v, N = l16
    if (fullRows) {
        #pragma unroll
        for (int t = 0; t < 4; ++t) {
            const int gcol = colOff + cw * 64 + t * 16 + l16;
            float* cp = C + (long)(rowBase + rw * 16 + half * 8) * ldc + gcol;
            #pragma unroll
            for (int v = 0; v < 8; ++v) cp[(long)v * ldc] = acc[t][v];
        }
    } else {
        #pragma unroll
        for (int t = 0; t < 4; ++t) {
            const int gcol = colOff + cw * 64 + t * 16 + l16;
            #pragma unroll
            for (int v = 0; v < 8; ++v) {
                int gr = rowBase + rw * 16 + half * 8 + v;
                if (gr < n) C[(long)gr * ldc + gcol] = acc[t][v];
            }
        }
    }
}

// ---------------------------------------------------------------------
// out[b,nn,d] = relu(bias[d] + sum_{h=0..127} H[nn, map(h,b)] * W[h,d]) (+xin)
// H row layout: support k at cols k*128 + b*64 + d'   (ldH = 256)
// ---------------------------------------------------------------------
__global__ __launch_bounds__(256)
void linear_gc(const float* __restrict__ H, const float* __restrict__ W,
               const float* __restrict__ bias, const float* __restrict__ xin,
               float* __restrict__ out, int n) {
    __shared__ float Wl[128 * 64];      // 32 KB
    __shared__ float Hs[4][128];
    const int tid = threadIdx.x;
    for (int i = tid; i < 128 * 64; i += 256) Wl[i] = W[i];
    const int grp = tid >> 6, d = tid & 63;
    const long base = (long)blockIdx.x * 4 + grp;   // index over B*n rows
    const bool ok = base < 2L * n;
    const int b  = ok ? (int)(base / n) : 0;
    const int nn = ok ? (int)(base % n) : 0;
    for (int j = d; j < 128; j += 64) {
        int k = j >> 6, dp = j & 63;
        Hs[grp][j] = ok ? H[(long)nn * 256 + k * 128 + b * 64 + dp] : 0.f;
    }
    __syncthreads();
    if (!ok) return;
    float acc = bias[d];
    #pragma unroll 8
    for (int h = 0; h < 128; ++h) acc = fmaf(Hs[grp][h], Wl[h * 64 + d], acc);
    acc = fmaxf(acc, 0.f);
    const long oidx = (long)b * n * DD + (long)nn * DD + d;
    if (xin) acc += xin[oidx];
    out[oidx] = acc;
}

// ---------------------------------------------------------------------
// z = sigmoid(t@Wt + s@Ws + b);  x += z*t + (1-z)*s
// ---------------------------------------------------------------------
__global__ __launch_bounds__(256)
void fusion(const float* __restrict__ t, const float* __restrict__ s,
            const float* __restrict__ Wt, const float* __restrict__ Ws,
            const float* __restrict__ bias, float* __restrict__ x, int n) {
    __shared__ float WtS[64 * 64], WsS[64 * 64];   // 16 KB each
    __shared__ float ts[4][64], ss[4][64];
    const int tid = threadIdx.x;
    for (int i = tid; i < 64 * 64; i += 256) { WtS[i] = Wt[i]; WsS[i] = Ws[i]; }
    const int grp = tid >> 6, d = tid & 63;
    const long base = (long)blockIdx.x * 4 + grp;
    const bool ok = base < 2L * n;
    const int b  = ok ? (int)(base / n) : 0;
    const int nn = ok ? (int)(base % n) : 0;
    const long oidx = (long)b * n * DD + (long)nn * DD + d;
    ts[grp][d] = ok ? t[oidx] : 0.f;
    ss[grp][d] = ok ? s[oidx] : 0.f;
    __syncthreads();
    if (!ok) return;
    float acc = bias[d];
    #pragma unroll 8
    for (int dp = 0; dp < 64; ++dp)
        acc += ts[grp][dp] * WtS[dp * 64 + d] + ss[grp][dp] * WsS[dp * 64 + d];
    const float z = 1.f / (1.f + expf(-acc));
    x[oidx] += z * ts[grp][d] + (1.f - z) * ss[grp][d];
}

// ---------------------------------------------------------------------
extern "C" void kernel_launch(void* const* d_in, const int* in_sizes, int n_in,
                              void* d_out, int out_size, void* d_ws, size_t ws_size,
                              hipStream_t stream) {
    const float* A0      = (const float*)d_in[0];
    const float* A1      = (const float*)d_in[1];
    const float* A2      = (const float*)d_in[2];
    const float* x0      = (const float*)d_in[3];
    const float* x1      = (const float*)d_in[4];
    const float* src_nv1 = (const float*)d_in[5];
    const float* src_nv2 = (const float*)d_in[6];
    const float* src_W   = (const float*)d_in[7];
    const float* src_b   = (const float*)d_in[8];
    const float* vna_nv1 = (const float*)d_in[9];
    const float* vna_nv2 = (const float*)d_in[10];
    const float* vna_W   = (const float*)d_in[11];
    const float* vna_b   = (const float*)d_in[12];
    const float* tgt_nv1 = (const float*)d_in[13];
    const float* tgt_nv2 = (const float*)d_in[14];
    const float* tgt_W   = (const float*)d_in[15];
    const float* tgt_b   = (const float*)d_in[16];
    const float* fus_Wt  = (const float*)d_in[17];
    const float* fus_Ws  = (const float*)d_in[18];
    const float* fus_b   = (const float*)d_in[19];
    float* out = (float*)d_out;

    // workspace carve-out (256B aligned)
    char* w = (char*)d_ws;
    auto carve = [&](size_t bytes) -> void* {
        void* p = (void*)w;
        w += (bytes + 255) & ~(size_t)255;
        return p;
    };
    unsigned short* adpS = (unsigned short*)carve((size_t)NS * NS * 2);   // 50 MB
    unsigned short* adpV = (unsigned short*)carve((size_t)NT * NS * 2);   // 40 MB
    unsigned short* adpT = (unsigned short*)carve((size_t)NT * NT * 2);   // 32 MB
    unsigned short* XTs  = (unsigned short*)carve((size_t)128 * KPS * 2);
    unsigned short* XTt  = (unsigned short*)carve((size_t)128 * KPT * 2);
    float* H    = (float*)carve((size_t)NS * 256 * 4);
    float* xsrc = (float*)carve((size_t)2 * NS * DD * 4);
    float* semb = (float*)carve((size_t)3 * 2 * NT * DD * 4);
    float* tbuf = (float*)carve((size_t)2 * NT * DD * 4);

    // adaptive adjacencies -> bf16
    adp_kernel<<<NS, 256, 0, stream>>>(src_nv1, src_nv2, adpS, NS, NS);
    adp_kernel<<<NT, 256, 0, stream>>>(vna_nv1, vna_nv2, adpV, NT, NS);
    adp_kernel<<<NT, 256, 0, stream>>>(tgt_nv1, tgt_nv2, adpT, NT, NT);

    // running states
    copyf<<<1024, 256, 0, stream>>>(x1, xsrc, 2L * NS * DD);
    copyf<<<1024, 256, 0, stream>>>(x0, out,  2L * NT * DD);

    const dim3 gS((NS + 63) / 64), gT((NT + 63) / 64);

    // ---- source + VNA blocks (interleaved so layer_results never stored) ----
    for (int i = 0; i < 3; ++i) {
        pack_xt<<<1024, 256, 0, stream>>>(xsrc, XTs, NS, KPS);
        // VNA: source_embs[i] = gc([A1, adpV], layer_results[i])
        gemm_bf16<true ><<<gT, 256, 0, stream>>>(A1,   XTs, H, NT, NS, KPS, 256, 0);
        gemm_bf16<false><<<gT, 256, 0, stream>>>(adpV, XTs, H, NT, NS, KPS, 256, 128);
        linear_gc<<<(2 * NT) / 4, 256, 0, stream>>>(H, vna_W + (size_t)i * 128 * 64,
                                                    vna_b + i * 64, nullptr,
                                                    semb + (size_t)i * 2 * NT * DD, NT);
        if (i < 2) {
            // source residual layer: x += gc([A2, adpS], x)
            gemm_bf16<true ><<<gS, 256, 0, stream>>>(A2,   XTs, H, NS, NS, KPS, 256, 0);
            gemm_bf16<false><<<gS, 256, 0, stream>>>(adpS, XTs, H, NS, NS, KPS, 256, 128);
            linear_gc<<<(2 * NS) / 4, 256, 0, stream>>>(H, src_W + (size_t)i * 128 * 64,
                                                        src_b + i * 64, xsrc, xsrc, NS);
        }
    }

    // ---- target block with gated fusion ----
    for (int i = 0; i < 3; ++i) {
        pack_xt<<<1024, 256, 0, stream>>>(out, XTt, NT, KPT);
        gemm_bf16<true ><<<gT, 256, 0, stream>>>(A0,   XTt, H, NT, NT, KPT, 256, 0);
        gemm_bf16<false><<<gT, 256, 0, stream>>>(adpT, XTt, H, NT, NT, KPT, 256, 128);
        linear_gc<<<(2 * NT) / 4, 256, 0, stream>>>(H, tgt_W + (size_t)i * 128 * 64,
                                                    tgt_b + i * 64, nullptr, tbuf, NT);
        fusion<<<(2 * NT) / 4, 256, 0, stream>>>(tbuf, semb + (size_t)i * 2 * NT * DD,
                                                 fus_Wt + (size_t)i * 64 * 64,
                                                 fus_Ws + (size_t)i * 64 * 64,
                                                 fus_b + i * 64, out, NT);
    }
}
